// GRAM_3478923509911
// MI455X (gfx1250) — matrix operand order repaired and checked by reading, not compiled, not run on hardware
//
#include <hip/hip_runtime.h>
#include <hip/hip_bf16.h>

// ---------------------------------------------------------------------------
// CDNA5 / gfx1250 implementation. Wave32, WMMA bf16 (16x16x32) for all GEMMs,
// TDM (tensor_load_to_lds) double-buffered A-tile staging in the GEMM.
// ---------------------------------------------------------------------------

typedef __attribute__((ext_vector_type(16))) __bf16 v16bf;
typedef __attribute__((ext_vector_type(8)))  float  v8f;
typedef __attribute__((ext_vector_type(4)))  unsigned int u32x4;
typedef __attribute__((ext_vector_type(8)))  int   i32x8;
typedef __attribute__((ext_vector_type(4)))  int   i32x4;

#define Dm 512
#define DRm 128
#define Pm 96
#define Km 5
#define Mm 4096
#define Bm 32
#define Lm 512
#define Hm 4
#define DHm 128
#define TEMPm 0.1f

#if __has_builtin(__builtin_amdgcn_tensor_load_to_lds) && \
    __has_builtin(__builtin_amdgcn_s_wait_tensorcnt)
#define USE_TDM 1
#else
#define USE_TDM 0
#endif

static __device__ __forceinline__ float gelu_f(float x) {
    return 0.5f * x * (1.f + erff(x * 0.70710678118654752f));
}

// ===========================================================================
// Generic WMMA GEMM: Y[r,n] = act( sum_k A[r,k] * W[n,k] + bias[n] )
// A: rows via aRowStride (elements). W: (N,Kd) row-major. Y: (R,N) row-major.
// block = 256 threads = 8 waves; wave w computes tile (blockIdx.x, blockIdx.y*8+w).
// A tile (16 x 64 f32) staged in LDS (TDM async when available), double-buffered.
// Requires: R %16==0, N %128==0, Kd %64==0.
// ===========================================================================
#define KSTEP 64

__global__ __launch_bounds__(256) void GRAM_gemm(
    const float* __restrict__ A, long aRowStride,
    const float* __restrict__ W, const float* __restrict__ bias,
    float* __restrict__ Y, int R, int N, int Kd, int act)
{
    __shared__ float As[2][16 * KSTEP];
    const int t    = threadIdx.x;
    const int lane = t & 31;
    const int wv   = t >> 5;
    const int rt   = blockIdx.x;
    const int nt   = blockIdx.y * 8 + wv;
    const int  l16  = lane & 15;
    const bool lohf = (lane < 16);

    const float* wr = W + (long)(nt * 16 + l16) * Kd;   // B column n = lane%16

#if USE_TDM
    const unsigned long long abase =
        (unsigned long long)(const void*)(A + (long)(rt * 16) * aRowStride);
    const unsigned long long strideE = (unsigned long long)aRowStride;  // elements
    auto issue_tdm = [&](int kb, int buf) {
        unsigned long long ga = abase + (unsigned long long)kb * 4ull;
        unsigned lds = (unsigned)(unsigned long long)(const void*)&As[buf][0];
        u32x4 g0;
        g0[0] = 1u;                                           // count=1, user mode
        g0[1] = lds;                                          // lds_addr (bytes)
        g0[2] = (unsigned)ga;                                 // global_addr[31:0]
        g0[3] = (unsigned)((ga >> 32) & 0x1FFFFFFull) | (2u << 30);  // addr hi | type=2
        i32x8 g1;
        const unsigned long long td0 = (unsigned long long)Kd;  // tensor_dim0
        const unsigned long long td1 = 16ull;                   // tensor_dim1
        g1[0] = (int)(2u << 16);                              // data_size = 4 bytes
        g1[1] = (int)((td0 & 0xFFFFull) << 16);               // tensor_dim0 lo
        g1[2] = (int)(((td0 >> 16) & 0xFFFFull) | ((td1 & 0xFFFFull) << 16));
        g1[3] = (int)(((td1 >> 16) & 0xFFFFull) | ((unsigned)KSTEP << 16)); // tile_dim0
        g1[4] = (int)16;                                      // tile_dim1 (tile_dim2=0)
        g1[5] = (int)(unsigned)strideE;                       // dim0 stride lo
        g1[6] = (int)(unsigned)((strideE >> 32) & 0xFFFFull); // dim0 stride hi
        g1[7] = 0;
        i32x4 z4 = {0, 0, 0, 0};
        i32x8 z8 = {0, 0, 0, 0, 0, 0, 0, 0};
        __builtin_amdgcn_tensor_load_to_lds(g0, g1, z4, z4, z8, 0);
    };
    if (wv == 0) issue_tdm(0, 0);
#else
    {
        int idx = t * 4, row = idx >> 6, col = idx & 63;      // 16x64 cooperative
        float4 f = *(const float4*)(A + (long)(rt * 16 + row) * aRowStride + col);
        *(float4*)&As[0][row * KSTEP + col] = f;
    }
#endif

    v8f acc = {};
    int buf = 0;
    for (int kb = 0; kb < Kd; kb += KSTEP, buf ^= 1) {
#if USE_TDM
        if (wv == 0) {
            if (kb + KSTEP < Kd) {
                issue_tdm(kb + KSTEP, buf ^ 1);               // prefetch next tile
                __builtin_amdgcn_s_wait_tensorcnt(1);         // oldest (current) done
            } else {
                __builtin_amdgcn_s_wait_tensorcnt(0);
            }
        }
        __syncthreads();                                      // As[buf] ready for all
#else
        __syncthreads();                                      // As[buf] visible
        if (kb + KSTEP < Kd) {
            int idx = t * 4, row = idx >> 6, col = idx & 63;
            float4 f = *(const float4*)(A + (long)(rt * 16 + row) * aRowStride
                                        + kb + KSTEP + col);
            *(float4*)&As[buf ^ 1][row * KSTEP + col] = f;
        }
#endif
        if (kb + KSTEP < Kd) __builtin_prefetch(wr + kb + KSTEP, 0, 1);

        #pragma unroll
        for (int c = 0; c < 2; ++c) {
            v16bf aa, bb;
            #pragma unroll
            for (int v = 0; v < 8; ++v) {
                // A 16x32 bf16 layout: lane<16 -> K {0..7,16..23}; lane>=16 -> +8
                int ka = ((v & 3) * 2) + ((v >> 2) * 16) + (lohf ? 0 : 8);
                float2 fa = *(const float2*)&As[buf][l16 * KSTEP + c * 32 + ka];
                aa[2 * v]     = (__bf16)fa.x;
                aa[2 * v + 1] = (__bf16)fa.y;
                // B 32x16 bf16 layout: same K order as A (lane<16 -> K {0..7,16..23}; lane>=16 -> +8)
                int kk = ka;
                float2 fb = *(const float2*)(wr + kb + c * 32 + kk);
                bb[2 * v]     = (__bf16)fb.x;
                bb[2 * v + 1] = (__bf16)fb.y;
            }
            acc = __builtin_amdgcn_wmma_f32_16x16x32_bf16(
                false, aa, false, bb, (short)0, acc, false, false);
        }
        __syncthreads();   // all reads of As[buf] done before it is overwritten
    }

    const int nc = nt * 16 + l16;                // C-layout: N = lane%16
    if (nc < N) {
        const float bv = bias ? bias[nc] : 0.f;
        #pragma unroll
        for (int v = 0; v < 8; ++v) {
            int Mr = v + (lohf ? 0 : 8);         // C-layout: M = vgpr + 8*half
            float y = acc[v] + bv;
            if (act == 1) y = gelu_f(y);
            Y[(long)(rt * 16 + Mr) * N + nc] = y;
        }
    }
}

// ===========================================================================
// Fused multi-head attention, one wave per (batch, head, 16-query tile).
// Scores strip (16 x <=544) lives in LDS. WMMA for Q.K^T and P.V.
// ===========================================================================
#define SC_COLS 544

__global__ __launch_bounds__(32) void GRAM_attn(
    const float* __restrict__ qP, long qBatch, int qRow,
    const float* __restrict__ kP, long kBatch, int kRow,
    const float* __restrict__ vP, long vBatch, int vRow,
    float* __restrict__ oP, long oBatch, int oRow,
    int qlen, int klen, float scale)
{
    __shared__ float sc[16 * SC_COLS];
    const int lane = threadIdx.x;
    const int qt = blockIdx.x, h = blockIdx.y, b = blockIdx.z;
    const int  l16  = lane & 15;
    const bool lohf = (lane < 16);

    const float* qBase = qP + (long)b * qBatch + h * DHm;
    const float* kBase = kP + (long)b * kBatch + h * DHm;
    const float* vBase = vP + (long)b * vBatch + h * DHm;

    // ---- load Q tile (16 x 128) as 4 A-layout chunks --------------------
    int qrow = qt * 16 + l16; if (qrow >= qlen) qrow = qlen - 1;
    const float* qr = qBase + (long)qrow * qRow;
    v16bf qa[4];
    #pragma unroll
    for (int c = 0; c < 4; ++c) {
        #pragma unroll
        for (int v = 0; v < 8; ++v) {
            int ka = ((v & 3) * 2) + ((v >> 2) * 16) + (lohf ? 0 : 8);
            float2 f = *(const float2*)(qr + c * 32 + ka);
            qa[c][2 * v]     = (__bf16)f.x;
            qa[c][2 * v + 1] = (__bf16)f.y;
        }
    }

    const int KTP = ((klen + 31) / 32) * 2;      // 16-wide key tiles, even count

    // ---- pass 1: scores -> LDS ------------------------------------------
    for (int kt = 0; kt < KTP; ++kt) {
        int key  = kt * 16 + l16;
        int keyc = key < klen ? key : klen - 1;
        const float* kr = kBase + (long)keyc * kRow;
        v8f acc = {};
        #pragma unroll
        for (int c = 0; c < 4; ++c) {
            v16bf bb;
            #pragma unroll
            for (int v = 0; v < 8; ++v) {
                int kk = ((v & 3) * 2) + ((v >> 2) * 16) + (lohf ? 0 : 8);
                float2 f = *(const float2*)(kr + c * 32 + kk);
                bb[2 * v]     = (__bf16)f.x;
                bb[2 * v + 1] = (__bf16)f.y;
            }
            acc = __builtin_amdgcn_wmma_f32_16x16x32_bf16(
                false, qa[c], false, bb, (short)0, acc, false, false);
        }
        #pragma unroll
        for (int v = 0; v < 8; ++v) {
            int Mr = v + (lohf ? 0 : 8);
            sc[Mr * SC_COLS + kt * 16 + l16] =
                (key < klen) ? acc[v] * scale : -1e30f;
        }
    }
    __syncthreads();

    // ---- pass 2: softmax (lane r handles row r) -------------------------
    if (lane < 16) {
        const int n = KTP * 16;
        float* rowp = &sc[lane * SC_COLS];
        float m = -1e30f;
        for (int j = 0; j < n; ++j) m = fmaxf(m, rowp[j]);
        float s = 0.f;
        for (int j = 0; j < n; ++j) { float e = __expf(rowp[j] - m); rowp[j] = e; s += e; }
        float inv = 1.f / s;
        for (int j = 0; j < n; ++j) rowp[j] *= inv;
    }
    __syncthreads();

    // ---- pass 3: out = P @ V  (8 column chunks of 16 over dh=128) ------
    for (int cb = 0; cb < DHm / 16; ++cb) {
        v8f acc = {};
        for (int kt2 = 0; kt2 < KTP / 2; ++kt2) {
            const int kbase = kt2 * 32;
            v16bf pa, vb;
            #pragma unroll
            for (int v = 0; v < 8; ++v) {
                // A = P chunk (16 x 32) from LDS, A-layout
                int ka = ((v & 3) * 2) + ((v >> 2) * 16) + (lohf ? 0 : 8);
                pa[2 * v]     = (__bf16)sc[l16 * SC_COLS + kbase + ka];
                pa[2 * v + 1] = (__bf16)sc[l16 * SC_COLS + kbase + ka + 1];
                // B = V chunk (32 keys x 16 cols), B-layout, rows strided
                int kk = ka;
                int k0 = kbase + kk;     int k0c = k0 < klen ? k0 : klen - 1;
                int k1 = k0 + 1;         int k1c = k1 < klen ? k1 : klen - 1;
                vb[2 * v]     = (__bf16)vBase[(long)k0c * vRow + cb * 16 + l16];
                vb[2 * v + 1] = (__bf16)vBase[(long)k1c * vRow + cb * 16 + l16];
            }
            acc = __builtin_amdgcn_wmma_f32_16x16x32_bf16(
                false, pa, false, vb, (short)0, acc, false, false);
        }
        #pragma unroll
        for (int v = 0; v < 8; ++v) {
            int Mr = v + (lohf ? 0 : 8);
            int q  = qt * 16 + Mr;
            if (q < qlen)
                oP[(long)b * oBatch + (long)q * oRow + h * DHm + cb * 16 + l16] = acc[v];
        }
    }
}

// ===========================================================================
// LayerNorm with optional residual and optional GELU: Y = act(LN(X (+Rsd)))
// one block (256 threads) per row.
// ===========================================================================
__global__ __launch_bounds__(256) void GRAM_ln(
    const float* __restrict__ X, const float* __restrict__ Rsd,
    const float* __restrict__ g, const float* __restrict__ bta,
    float* __restrict__ Y, int Dn, int act)
{
    const int row = blockIdx.x, t = threadIdx.x;
    const float* xr = X + (long)row * Dn;
    const float* rr = Rsd ? Rsd + (long)row * Dn : nullptr;
    float s = 0.f, s2 = 0.f;
    for (int j = t; j < Dn; j += 256) {
        float v = xr[j] + (rr ? rr[j] : 0.f);
        s += v; s2 += v * v;
    }
    __shared__ float rs[8], rs2[8];
    for (int o = 16; o > 0; o >>= 1) { s += __shfl_down(s, o, 32); s2 += __shfl_down(s2, o, 32); }
    if ((t & 31) == 0) { rs[t >> 5] = s; rs2[t >> 5] = s2; }
    __syncthreads();
    __shared__ float mean_s, inv_s;
    if (t == 0) {
        float S = 0.f, S2 = 0.f;
        for (int w = 0; w < 8; ++w) { S += rs[w]; S2 += rs2[w]; }
        float m = S / Dn;
        mean_s = m;
        inv_s = rsqrtf(S2 / Dn - m * m + 1e-5f);
    }
    __syncthreads();
    const float m = mean_s, inv = inv_s;
    for (int j = t; j < Dn; j += 256) {
        float v = xr[j] + (rr ? rr[j] : 0.f);
        float y = (v - m) * inv * g[j] + bta[j];
        if (act == 1) y = gelu_f(y);
        Y[(long)row * Dn + j] = y;
    }
}

// ---- build h = concat(cls, x) over sequence dim ---------------------------
__global__ void GRAM_buildh(const float* __restrict__ x, const float* __restrict__ cls,
                            float* __restrict__ h, int n)
{
    int i = blockIdx.x * blockDim.x + threadIdx.x;
    if (i >= n) return;
    int d = i % Dm; long bl = i / Dm; int l = (int)(bl % (Lm + 1)); long b = bl / (Lm + 1);
    h[i] = (l == 0) ? cls[d] : x[(b * Lm + (l - 1)) * Dm + d];
}

// ---- mean over sequence: (B,L,D) -> (B,D) ---------------------------------
__global__ void GRAM_mean1(const float* __restrict__ X, float* __restrict__ Y, int n)
{
    int i = blockIdx.x * blockDim.x + threadIdx.x;
    if (i >= n) return;
    int d = i % Dm; int b = i / Dm;
    const float* p = X + (long)b * Lm * Dm + d;
    float s = 0.f;
    for (int l = 0; l < Lm; ++l) s += p[(long)l * Dm];
    Y[i] = s * (1.f / Lm);
}

// ---- q_key = sw*q_repr + (1-sw)*q_val -------------------------------------
__global__ void GRAM_mix(const float* __restrict__ a, const float* __restrict__ b2,
                         const float* __restrict__ sw, float* __restrict__ y, int n)
{
    int i = blockIdx.x * blockDim.x + threadIdx.x;
    if (i >= n) return;
    float s = sw[0];
    y[i] = s * a[i] + (1.f - s) * b2[i];
}

// ---- distances + top-K=5 + softmax weights, one block per batch -----------
__global__ __launch_bounds__(256) void GRAM_topk(
    const float* __restrict__ qkey, const float* __restrict__ mk,
    float* __restrict__ simw, int* __restrict__ topidx)
{
    __shared__ float dist[Mm];
    __shared__ float rv[256]; __shared__ int ri[256];
    __shared__ float sel[Km]; __shared__ int seli[Km];
    const int b = blockIdx.x, t = threadIdx.x;
    const float* q = qkey + b * DRm;
    for (int m = t; m < Mm; m += 256) {
        const float* mm = mk + (long)m * DRm;
        float s = 0.f;
        for (int d = 0; d < DRm; ++d) { float df = q[d] - mm[d]; s += df * df; }
        dist[m] = s;
    }
    __syncthreads();
    for (int kk = 0; kk < Km; ++kk) {
        float best = 3.4e38f; int bi = 0;
        for (int m = t; m < Mm; m += 256)
            if (dist[m] < best) { best = dist[m]; bi = m; }
        rv[t] = best; ri[t] = bi;
        __syncthreads();
        for (int o = 128; o > 0; o >>= 1) {
            if (t < o && rv[t + o] < rv[t]) { rv[t] = rv[t + o]; ri[t] = ri[t + o]; }
            __syncthreads();
        }
        if (t == 0) { sel[kk] = rv[0]; seli[kk] = ri[0]; dist[ri[0]] = 3.4e38f; }
        __syncthreads();
    }
    if (t == 0) {
        float scv[Km], mx = -3.4e38f;
        for (int kk = 0; kk < Km; ++kk) { scv[kk] = -sel[kk] / TEMPm; mx = fmaxf(mx, scv[kk]); }
        float s = 0.f;
        for (int kk = 0; kk < Km; ++kk) { scv[kk] = __expf(scv[kk] - mx); s += scv[kk]; }
        for (int kk = 0; kk < Km; ++kk) { simw[b * Km + kk] = scv[kk] / s; topidx[b * Km + kk] = seli[kk]; }
    }
}

// ---- ret_pooled = memory_values[idx].mean(P) ------------------------------
__global__ void GRAM_refmean(const float* __restrict__ MV, const int* __restrict__ idx,
                             float* __restrict__ Y, int n)
{
    int i = blockIdx.x * blockDim.x + threadIdx.x;
    if (i >= n) return;
    int d = i % Dm; int bk = i / Dm;
    const float* p = MV + (long)idx[bk] * Pm * Dm + d;
    float s = 0.f;
    for (int pp = 0; pp < Pm; ++pp) s += p[(long)pp * Dm];
    Y[i] = s * (1.f / Pm);
}

// ---- mi = concat(cur_enc broadcast, ret_enc) on last dim ------------------
__global__ void GRAM_concat_mi(const float* __restrict__ ce, const float* __restrict__ re,
                               float* __restrict__ y, int n)
{
    int i = blockIdx.x * blockDim.x + threadIdx.x;
    if (i >= n) return;
    int c = i % (2 * Dm); int bk = i / (2 * Dm); int b = bk / Km;
    y[i] = (c < Dm) ? ce[b * Dm + c] : re[(long)bk * Dm + (c - Dm)];
}

// ---- gi = concat(cur_enc, wref_enc) ---------------------------------------
__global__ void GRAM_concat2(const float* __restrict__ a, const float* __restrict__ b2,
                             float* __restrict__ y, int n)
{
    int i = blockIdx.x * blockDim.x + threadIdx.x;
    if (i >= n) return;
    int c = i % (2 * Dm); int r = i / (2 * Dm);
    y[i] = (c < Dm) ? a[r * Dm + c] : b2[r * Dm + (c - Dm)];
}

// ---- row-dot + sigmoid: y[row] = sigmoid(X[row,:].w + b) ------------------
__global__ __launch_bounds__(256) void GRAM_dot_sig(
    const float* __restrict__ X, const float* __restrict__ w,
    const float* __restrict__ bb, float* __restrict__ Y, int Dn)
{
    const int row = blockIdx.x, t = threadIdx.x;
    const float* x = X + (long)row * Dn;
    float s = 0.f;
    for (int j = t; j < Dn; j += 256) s += x[j] * w[j];
    __shared__ float red[8];
    for (int o = 16; o > 0; o >>= 1) s += __shfl_down(s, o, 32);
    if ((t & 31) == 0) red[t >> 5] = s;
    __syncthreads();
    if (t == 0) {
        float S = 0.f;
        for (int wv = 0; wv < 8; ++wv) S += red[wv];
        Y[row] = 1.f / (1.f + __expf(-(S + bb[0])));
    }
}

// ---- cw = normalize(simw * match) -----------------------------------------
__global__ void GRAM_cw(const float* __restrict__ simw, const float* __restrict__ match,
                        float* __restrict__ cw_)
{
    int b = blockIdx.x * blockDim.x + threadIdx.x;
    if (b >= Bm) return;
    float tv[Km], s = 0.f;
    for (int k = 0; k < Km; ++k) { tv[k] = simw[b * Km + k] * match[b * Km + k]; s += tv[k]; }
    float inv = 1.f / (s + 1e-8f);
    for (int k = 0; k < Km; ++k) cw_[b * Km + k] = tv[k] * inv;
}

// ---- wref = einsum('bk,bkd->bd', cw, ret_pooled) --------------------------
__global__ void GRAM_wsum(const float* __restrict__ cw_, const float* __restrict__ rp,
                          float* __restrict__ Y, int n)
{
    int i = blockIdx.x * blockDim.x + threadIdx.x;
    if (i >= n) return;
    int d = i % Dm; int b = i / Dm;
    float s = 0.f;
    for (int k = 0; k < Km; ++k) s += cw_[b * Km + k] * rp[((long)b * Km + k) * Dm + d];
    Y[i] = s;
}

// ---- fused = gate*x + (1-gate)*attn_out; write concat [x, fused] ----------
__global__ void GRAM_fuse(const float* __restrict__ x, const float* __restrict__ ao,
                          const float* __restrict__ gate, float* __restrict__ cat, int n)
{
    int i = blockIdx.x * blockDim.x + threadIdx.x;
    if (i >= n) return;
    int d = i % Dm; long bl = i / Dm; int b = (int)(bl / Lm);
    float g = gate[b];
    float f = g * x[i] + (1.f - g) * ao[i];
    float* row = cat + bl * (2 * Dm);
    row[d] = x[i];
    row[Dm + d] = f;
}

// ===========================================================================
// Host-side launch
// ===========================================================================
static inline dim3 g1(long n) { return dim3((unsigned)((n + 255) / 256)); }

extern "C" void kernel_launch(void* const* d_in, const int* in_sizes, int n_in,
                              void* d_out, int out_size, void* d_ws, size_t ws_size,
                              hipStream_t stream)
{
    (void)in_sizes; (void)n_in; (void)out_size; (void)ws_size;
    int ii = 0;
    auto F = [&](int idx) { return (const float*)d_in[idx]; };
    const float* x   = F(ii++);                 // (B,L,D)
    const float* cls = F(ii++);                 // pe.cls
    struct Layer { const float *wi,*bi,*wo,*bo,*l1g,*l1b,*w1,*b1,*w2,*b2,*l2g,*l2b; } Lr[2];
    for (int l = 0; l < 2; ++l) {
        Lr[l].wi = F(ii++); Lr[l].bi = F(ii++); Lr[l].wo = F(ii++); Lr[l].bo = F(ii++);
        Lr[l].l1g = F(ii++); Lr[l].l1b = F(ii++);
        Lr[l].w1 = F(ii++); Lr[l].b1 = F(ii++); Lr[l].w2 = F(ii++); Lr[l].b2 = F(ii++);
        Lr[l].l2g = F(ii++); Lr[l].l2b = F(ii++);
    }
    const float *pw = F(ii++), *pb = F(ii++), *plg = F(ii++), *plb = F(ii++);
    const float *vw1 = F(ii++), *vb1 = F(ii++), *vlg = F(ii++), *vlb = F(ii++),
                *vw2 = F(ii++), *vb2 = F(ii++);
    const float *shapew = F(ii++);
    const float *cwW = F(ii++), *cbW = F(ii++), *clg = F(ii++), *clb = F(ii++),
                *rw = F(ii++), *rb = F(ii++), *rlg = F(ii++), *rlb = F(ii++),
                *msw1 = F(ii++), *msb1 = F(ii++), *msw2 = F(ii++), *msb2 = F(ii++),
                *gw1 = F(ii++), *gb1 = F(ii++), *gw2 = F(ii++), *gb2 = F(ii++),
                *cawi = F(ii++), *cabi = F(ii++), *cawo = F(ii++), *cabo = F(ii++),
                *ow = F(ii++), *ob = F(ii++);
    const float *cpw = F(ii++), *cpb = F(ii++);
    const float *mkeys = F(ii++);
    const float *mvals = F(ii++);

    // ---- workspace carve-out (f32) ----------------------------------------
    float* ws = (float*)d_ws;
    size_t off = 0;
    auto alloc = [&](size_t n) { float* p = ws + off; off += (n + 63) & ~(size_t)63; return p; };
    const long LP1 = Lm + 1;                       // 513
    float* bufH   = alloc((size_t)Bm * LP1 * Dm);          // hidden
    float* bufQKV = alloc((size_t)Bm * LP1 * 3 * Dm);      // qkv / cross-q
    float* bufA   = alloc((size_t)Bm * LP1 * Dm);          // attention out / fo
    float* bufT   = alloc((size_t)Bm * LP1 * Dm);          // proj temp
    float* bufF   = alloc((size_t)Bm * LP1 * 2 * Dm);      // ffn hidden / concat
    float* xmean  = alloc(Bm * Dm);
    float* t128a  = alloc(Bm * DRm);
    float* qrepr  = alloc(Bm * DRm);
    float* t128b  = alloc(Bm * DRm);
    float* qval   = alloc(Bm * DRm);
    float* qkey   = alloc(Bm * DRm);
    float* curlin = alloc(Bm * Dm);
    float* curenc = alloc(Bm * Dm);
    float* simw   = alloc(Bm * Km);
    int*   topidx = (int*)alloc(Bm * Km);
    float* retp   = alloc(Bm * Km * Dm);
    float* retlin = alloc(Bm * Km * Dm);
    float* retenc = alloc(Bm * Km * Dm);
    float* mi     = alloc((size_t)Bm * Km * 2 * Dm);
    float* msh    = alloc(Bm * Km * Dm);
    float* match  = alloc(Bm * Km);
    float* cwv    = alloc(Bm * Km);
    float* wref   = alloc(Bm * Dm);
    float* wreflin= alloc(Bm * Dm);
    float* wrefenc= alloc(Bm * Dm);
    float* gi     = alloc(Bm * 2 * Dm);
    float* gh     = alloc(Bm * Dm);
    float* gate   = alloc(Bm);
    float* kv     = alloc((size_t)Bm * Km * 2 * Dm);

    const float scale = 0.08838834764831845f;      // 1/sqrt(128)
    const long Renc = (long)Bm * LP1;              // 16416 rows
    const long Rseq = (long)Bm * Lm;               // 16384 rows

    // ---- 1. h = concat(cls, x) --------------------------------------------
    GRAM_buildh<<<g1(Renc * Dm), 256, 0, stream>>>(x, cls, bufH, (int)(Renc * Dm));

    // ---- 2. encoder layers ------------------------------------------------
    for (int l = 0; l < 2; ++l) {
        const Layer& P = Lr[l];
        GRAM_gemm<<<dim3(Renc / 16, (3 * Dm) / 128), 256, 0, stream>>>(
            bufH, Dm, P.wi, P.bi, bufQKV, (int)Renc, 3 * Dm, Dm, 0);
        GRAM_attn<<<dim3((LP1 + 15) / 16, Hm, Bm), 32, 0, stream>>>(
            bufQKV,            LP1 * 3L * Dm, 3 * Dm,
            bufQKV + Dm,       LP1 * 3L * Dm, 3 * Dm,
            bufQKV + 2 * Dm,   LP1 * 3L * Dm, 3 * Dm,
            bufA,              LP1 * (long)Dm, Dm,
            (int)LP1, (int)LP1, scale);
        GRAM_gemm<<<dim3(Renc / 16, Dm / 128), 256, 0, stream>>>(
            bufA, Dm, P.wo, P.bo, bufT, (int)Renc, Dm, Dm, 0);
        GRAM_ln<<<dim3((unsigned)Renc), 256, 0, stream>>>(
            bufH, bufT, P.l1g, P.l1b, bufH, Dm, 0);
        GRAM_gemm<<<dim3(Renc / 16, (2 * Dm) / 128), 256, 0, stream>>>(
            bufH, Dm, P.w1, P.b1, bufF, (int)Renc, 2 * Dm, Dm, 1);
        GRAM_gemm<<<dim3(Renc / 16, Dm / 128), 256, 0, stream>>>(
            bufF, 2 * Dm, P.w2, P.b2, bufT, (int)Renc, Dm, 2 * Dm, 0);
        GRAM_ln<<<dim3((unsigned)Renc), 256, 0, stream>>>(
            bufH, bufT, P.l2g, P.l2b, bufH, Dm, 0);
    }

    // ---- 3. pooled paths --------------------------------------------------
    GRAM_mean1<<<g1(Bm * Dm), 256, 0, stream>>>(x, xmean, Bm * Dm);

    // q_repr = LN(h[:,0] @ pw^T + pb)   (gather rows via stride 513*D)
    GRAM_gemm<<<dim3(Bm / 16, 1), 256, 0, stream>>>(
        bufH, LP1 * (long)Dm, pw, pb, t128a, Bm, DRm, Dm, 0);
    GRAM_ln<<<dim3(Bm), 256, 0, stream>>>(t128a, nullptr, plg, plb, qrepr, DRm, 0);

    // q_val = (gelu(LN(xmean @ vw1^T + vb1))) @ vw2^T + vb2
    GRAM_gemm<<<dim3(Bm / 16, 1), 256, 0, stream>>>(
        xmean, Dm, vw1, vb1, t128b, Bm, DRm, Dm, 0);
    GRAM_ln<<<dim3(Bm), 256, 0, stream>>>(t128b, nullptr, vlg, vlb, t128b, DRm, 1);
    GRAM_gemm<<<dim3(Bm / 16, 1), 256, 0, stream>>>(
        t128b, DRm, vw2, vb2, qval, Bm, DRm, DRm, 0);

    GRAM_mix<<<g1(Bm * DRm), 256, 0, stream>>>(qrepr, qval, shapew, qkey, Bm * DRm);

    // ---- 4. retrieval -----------------------------------------------------
    GRAM_topk<<<dim3(Bm), 256, 0, stream>>>(qkey, mkeys, simw, topidx);
    GRAM_refmean<<<g1((long)Bm * Km * Dm), 256, 0, stream>>>(
        mvals, topidx, retp, Bm * Km * Dm);

    // ret_enc = gelu(LN(ret_pooled @ rw^T + rb))
    GRAM_gemm<<<dim3((Bm * Km) / 16, Dm / 128), 256, 0, stream>>>(
        retp, Dm, rw, rb, retlin, Bm * Km, Dm, Dm, 0);
    GRAM_ln<<<dim3(Bm * Km), 256, 0, stream>>>(retlin, nullptr, rlg, rlb, retenc, Dm, 1);

    // cur_enc = gelu(LN(xmean @ cw^T + cb))
    GRAM_gemm<<<dim3(Bm / 16, Dm / 128), 256, 0, stream>>>(
        xmean, Dm, cwW, cbW, curlin, Bm, Dm, Dm, 0);
    GRAM_ln<<<dim3(Bm), 256, 0, stream>>>(curlin, nullptr, clg, clb, curenc, Dm, 1);

    // match head
    GRAM_concat_mi<<<g1((long)Bm * Km * 2 * Dm), 256, 0, stream>>>(
        curenc, retenc, mi, Bm * Km * 2 * Dm);
    GRAM_gemm<<<dim3((Bm * Km) / 16, Dm / 128), 256, 0, stream>>>(
        mi, 2 * Dm, msw1, msb1, msh, Bm * Km, Dm, 2 * Dm, 1);
    GRAM_dot_sig<<<dim3(Bm * Km), 256, 0, stream>>>(msh, msw2, msb2, match, Dm);
    GRAM_cw<<<g1(Bm), 256, 0, stream>>>(simw, match, cwv);

    // wref_enc = gelu(LN((cw . ret_pooled) @ rw^T + rb))
    GRAM_wsum<<<g1(Bm * Dm), 256, 0, stream>>>(cwv, retp, wref, Bm * Dm);
    GRAM_gemm<<<dim3(Bm / 16, Dm / 128), 256, 0, stream>>>(
        wref, Dm, rw, rb, wreflin, Bm, Dm, Dm, 0);
    GRAM_ln<<<dim3(Bm), 256, 0, stream>>>(wreflin, nullptr, rlg, rlb, wrefenc, Dm, 1);

    // gate
    GRAM_concat2<<<g1(Bm * 2 * Dm), 256, 0, stream>>>(curenc, wrefenc, gi, Bm * 2 * Dm);
    GRAM_gemm<<<dim3(Bm / 16, Dm / 128), 256, 0, stream>>>(
        gi, 2 * Dm, gw1, gb1, gh, Bm, Dm, 2 * Dm, 1);
    GRAM_dot_sig<<<dim3(Bm), 256, 0, stream>>>(gh, gw2, gb2, gate, Dm);

    // ---- 5. cross attention ----------------------------------------------
    // Q = x @ wq^T (+bq)     [wq = cawi rows 0..D)
    GRAM_gemm<<<dim3(Rseq / 16, Dm / 128), 256, 0, stream>>>(
        x, Dm, cawi, cabi, bufQKV, (int)Rseq, Dm, Dm, 0);
    // KV = ret_enc @ [wk;wv]^T  [cawi rows D..3D)
    GRAM_gemm<<<dim3((Bm * Km) / 16, (2 * Dm) / 128), 256, 0, stream>>>(
        retenc, Dm, cawi + (long)Dm * Dm, cabi + Dm, kv, Bm * Km, 2 * Dm, Dm, 0);
    GRAM_attn<<<dim3(Lm / 16, Hm, Bm), 32, 0, stream>>>(
        bufQKV,    (long)Lm * Dm, Dm,
        kv,        (long)Km * 2 * Dm, 2 * Dm,
        kv + Dm,   (long)Km * 2 * Dm, 2 * Dm,
        bufA,      (long)Lm * Dm, Dm,
        Lm, Km, scale);
    GRAM_gemm<<<dim3(Rseq / 16, Dm / 128), 256, 0, stream>>>(
        bufA, Dm, cawo, cabo, bufT, (int)Rseq, Dm, Dm, 0);

    // ---- 6. fuse + output projections ------------------------------------
    GRAM_fuse<<<g1(Rseq * Dm), 256, 0, stream>>>(x, bufT, gate, bufF, (int)(Rseq * Dm));
    GRAM_gemm<<<dim3(Rseq / 16, Dm / 128), 256, 0, stream>>>(
        bufF, 2 * Dm, ow, ob, bufA, (int)Rseq, Dm, 2 * Dm, 0);
    GRAM_gemm<<<dim3(Rseq / 16, Dm / 128), 256, 0, stream>>>(
        bufA, Dm, cpw, cpb, (float*)d_out, (int)Rseq, Dm, Dm, 0);
}
